// SwinBlock_30099130810839
// MI455X (gfx1250) — matrix operand order repaired and checked by reading, not compiled, not run on hardware
//
#include <hip/hip_runtime.h>
#include <hip/hip_bf16.h>
#include <math.h>
#include <stdint.h>

typedef __attribute__((ext_vector_type(16))) _Float16 v16h;
typedef __attribute__((ext_vector_type(8)))  _Float16 v8h;
typedef __attribute__((ext_vector_type(8)))  float    v8f;
typedef __attribute__((ext_vector_type(4)))  unsigned int v4u;
typedef __attribute__((ext_vector_type(4)))  int      v4i;
typedef __attribute__((ext_vector_type(8)))  int      v8i;

#define DIM   256
#define HID   512
#define NH    8
#define HD    32
#define WS    8
#define NTOK  64          // tokens per window
#define BATCH 8
#define HH    128
#define WW    128
#define HWSZ  (HH*WW)     // 16384
#define NWIN  (BATCH*(HH/WS)*(WW/WS))   // 2048
#define MROWS (NWIN*NTOK)               // 131072
#define SCALE 0.17677669529663687f      // 32^-0.5

#ifndef __has_builtin
#define __has_builtin(x) 0
#endif
#if __has_builtin(__builtin_amdgcn_tensor_load_to_lds) && \
    __has_builtin(__builtin_amdgcn_s_wait_tensorcnt)
#define USE_TDM 1
#else
#define USE_TDM 0
#endif

// ---------------- WMMA helpers (CDNA5 16x16x32 f16 -> f32) ----------------

__device__ __forceinline__ v8f wmma16(v16h a, v16h b, v8f c) {
  return __builtin_amdgcn_wmma_f32_16x16x32_f16(false, a, false, b,
                                                (short)0, c, false, false);
}

// A-operand: row-major [16 x 32] tile at `base` with row stride `ld` (halves).
__device__ __forceinline__ v16h load_a16(const _Float16* base, int ld) {
  int lane = threadIdx.x & 31;
  int m = lane & 15, ks = lane >> 4;
  const _Float16* p = base + m * ld + ks * 8;
  v8h lo = *(const v8h*)(p);
  v8h hi = *(const v8h*)(p + 16);
  return __builtin_shufflevector(lo, hi, 0,1,2,3,4,5,6,7,8,9,10,11,12,13,14,15);
}

// B-operand from a transposed source: B[k][n] = S[n][k], S row-major stride ld.
__device__ __forceinline__ v16h load_bt16(const _Float16* base, int ld) {
  int lane = threadIdx.x & 31;
  int n = lane & 15, kh = lane >> 4;
  const _Float16* p = base + n * ld + kh * 8;
  v8h lo = *(const v8h*)(p);
  v8h hi = *(const v8h*)(p + 16);
  return __builtin_shufflevector(lo, hi, 0,1,2,3,4,5,6,7,8,9,10,11,12,13,14,15);
}

// 64x32 register-tile GEMM core.
__device__ __forceinline__ void gemm_tile_64x32(const _Float16* __restrict__ A, int lda,
                                                const _Float16* __restrict__ Bt, int ldbt,
                                                int Ktot, v8f (&acc)[4][2]) {
  for (int k0 = 0; k0 < Ktot; k0 += 32) {
    v16h b0 = load_bt16(Bt + k0, ldbt);
    v16h b1 = load_bt16(Bt + 16 * ldbt + k0, ldbt);
#pragma unroll
    for (int i = 0; i < 4; ++i) {
      v16h a = load_a16(A + (size_t)(i * 16) * lda + k0, lda);
      acc[i][0] = wmma16(a, b0, acc[i][0]);
      acc[i][1] = wmma16(a, b1, acc[i][1]);
    }
  }
}

#if USE_TDM
// 1-D contiguous DMA of nelem f16 (nelem < 65536) global -> LDS via Tensor Data Mover.
// D# per CDNA5 ISA: group0 {count,lds,global,type=2},
// group1 {data_size=2B, tensor_dim0=tile_dim0=nelem, dims1/2 trivial}.
__device__ __forceinline__ void tdm_copy_f16(void* lds_ptr, const void* gptr, unsigned nelem) {
  unsigned long long ga = (unsigned long long)(uintptr_t)gptr;
  unsigned lds_off = (unsigned)(uintptr_t)lds_ptr;   // LDS aperture: low 32 bits = LDS addr
  v4u g0;
  g0[0] = 1u;                                        // count=1, user descriptor
  g0[1] = lds_off;                                   // lds_addr
  g0[2] = (unsigned)(ga & 0xFFFFFFFFu);              // global_addr[31:0]
  g0[3] = (unsigned)((ga >> 32) & 0x01FFFFFFu) | 0x80000000u;  // ga[56:32] | type=2
  v8i g1;
  g1[0] = (int)(1u << 16);                           // data_size=1 (2 bytes)
  g1[1] = (int)((nelem & 0xFFFFu) << 16);            // tensor_dim0[15:0]
  g1[2] = (int)(((nelem >> 16) & 0xFFFFu) | (1u << 16)); // tensor_dim0 hi | tensor_dim1=1
  g1[3] = (int)((nelem & 0xFFFFu) << 16);            // tile_dim0 = nelem
  g1[4] = 1;                                         // tile_dim1=1, tile_dim2=0
  g1[5] = (int)nelem;                                // tensor_dim0_stride lo
  g1[6] = 0;
  g1[7] = 0;
  v4i z = {0, 0, 0, 0};
#if __clang_major__ >= 23
  v8i z8 = {0, 0, 0, 0, 0, 0, 0, 0};
  __builtin_amdgcn_tensor_load_to_lds(g0, g1, z, z, z8, 0);
#else
  __builtin_amdgcn_tensor_load_to_lds(g0, g1, z, z, 0);
#endif
}
#endif

// ---------------- small prep kernels ----------------

__global__ void k_transpose_f16(const float* __restrict__ w, _Float16* __restrict__ wt,
                                int K, int N) {
  int o = blockIdx.x * 256 + threadIdx.x;
  if (o < K * N) {
    int n = o / K, k = o % K;
    wt[o] = (_Float16)w[k * N + n];
  }
}

__global__ void k_biasdense(const float* __restrict__ rpb, const int* __restrict__ rpi,
                            float* __restrict__ bd) {
  int i = blockIdx.x * 256 + threadIdx.x;     // 8*4096
  int h = i >> 12, mn = i & 4095;
  bd[i] = rpb[rpi[mn] * NH + h];
}

// ---------------- LN1 + window partition -> xw f16 [MROWS][256] ----------------

__global__ void __launch_bounds__(256) k_ln1(const float* __restrict__ x,
                                             const float* __restrict__ g,
                                             const float* __restrict__ bta,
                                             _Float16* __restrict__ xw) {
  __shared__ float sx[256][17];
  int b  = blockIdx.y;
  int p0 = blockIdx.x * 16;
  int t  = threadIdx.x;
#pragma unroll
  for (int cb = 0; cb < 256; cb += 16) {
    int c = cb + (t >> 4), pl = t & 15;
    sx[c][pl] = x[((size_t)(b * 256 + c)) * HWSZ + p0 + pl];
  }
  __syncthreads();
  int wave = t >> 5, lane = t & 31;
  for (int ti = 0; ti < 2; ++ti) {
    int tok = wave * 2 + ti;
    float vals[8], sum = 0.f;
#pragma unroll
    for (int j = 0; j < 8; ++j) { vals[j] = sx[lane + 32 * j][tok]; sum += vals[j]; }
    for (int off = 16; off >= 1; off >>= 1) sum += __shfl_xor(sum, off, 32);
    float mu = sum * (1.f / 256.f);
    float var = 0.f;
#pragma unroll
    for (int j = 0; j < 8; ++j) { float d = vals[j] - mu; var += d * d; }
    for (int off = 16; off >= 1; off >>= 1) var += __shfl_xor(var, off, 32);
    float rs = rsqrtf(var * (1.f / 256.f) + 1e-5f);
    int p  = p0 + tok;
    int hh = p >> 7, wc = p & 127;
    int wh = hh >> 3, iy = hh & 7, wwi = wc >> 3, ix = wc & 7;
    int win = (b * 16 + wh) * 16 + wwi;
    int tt  = iy * WS + ix;
    size_t row = (size_t)(win * NTOK + tt) * 256;
#pragma unroll
    for (int j = 0; j < 8; ++j) {
      int c = lane + 32 * j;
      xw[row + c] = (_Float16)((vals[j] - mu) * rs * g[c] + bta[c]);
    }
  }
}

// ---------------- GEMM: QKV, head-split epilogue (hoisted addressing) ----------------
// q,k stored [bh][t][d]; v stored TRANSPOSED [bh][d][t].

__global__ void __launch_bounds__(128) k_gemm_qkv(const _Float16* __restrict__ xw,
                                                  const _Float16* __restrict__ wqT,
                                                  const float* __restrict__ qkvb,
                                                  _Float16* __restrict__ q,
                                                  _Float16* __restrict__ k,
                                                  _Float16* __restrict__ vt) {
  int wave = threadIdx.x >> 5, lane = threadIdx.x & 31;
  int n0 = blockIdx.x * 128 + wave * 32;
  int m0 = blockIdx.y * 64;
  v8f acc[4][2] = {};
  gemm_tile_64x32(xw + (size_t)m0 * 256, 256, wqT + (size_t)n0 * 256, 256, 256, acc);
  int nl = lane & 15, hs = lane >> 4;
  int win = m0 >> 6;                 // block-uniform: m range never crosses a window
#pragma unroll
  for (int j = 0; j < 2; ++j) {
    int nn = n0 + j * 16 + nl;
    int which = nn >> 8, c256 = nn & 255, head = c256 >> 5, d = c256 & 31;
    float bias = qkvb[nn];
    size_t bh = (size_t)(win * NH + head) * (NTOK * HD);
    if (which == 0) {
      _Float16* dst = q + bh + d;
#pragma unroll
      for (int i = 0; i < 4; ++i)
#pragma unroll
        for (int r = 0; r < 8; ++r) {
          int t = i * 16 + hs * 8 + r;
          dst[t * HD] = (_Float16)((acc[i][j][r] + bias) * SCALE);
        }
    } else if (which == 1) {
      _Float16* dst = k + bh + d;
#pragma unroll
      for (int i = 0; i < 4; ++i)
#pragma unroll
        for (int r = 0; r < 8; ++r) {
          int t = i * 16 + hs * 8 + r;
          dst[t * HD] = (_Float16)(acc[i][j][r] + bias);
        }
    } else {
      _Float16* dst = vt + bh + d * NTOK;
#pragma unroll
      for (int i = 0; i < 4; ++i)
#pragma unroll
        for (int r = 0; r < 8; ++r) {
          int t = i * 16 + hs * 8 + r;
          dst[t] = (_Float16)(acc[i][j][r] + bias);
        }
    }
  }
}

// ---------------- Attention per (window, head): TDM-staged operands ----------------

__global__ void __launch_bounds__(128) k_attn(const _Float16* __restrict__ q,
                                              const _Float16* __restrict__ kbuf,
                                              const _Float16* __restrict__ vt,
                                              const float* __restrict__ biasd,
                                              _Float16* __restrict__ o) {
  __shared__ __align__(16) _Float16 Qs[NTOK * HD];   // [64 t][32 d]
  __shared__ __align__(16) _Float16 Ks[NTOK * HD];   // [64 t][32 d]
  __shared__ __align__(16) _Float16 Vs[HD * NTOK];   // [32 d][64 t]  (V^T)
  __shared__ __align__(16) _Float16 P[64][64];
  int bh = blockIdx.x, head = bh & 7, win = bh >> 3;
  const _Float16* qh  = q    + (size_t)bh * (NTOK * HD);
  const _Float16* kh  = kbuf + (size_t)bh * (NTOK * HD);
  const _Float16* vth = vt   + (size_t)bh * (NTOK * HD);
  const float* bb = biasd + head * 4096;
  int wave = threadIdx.x >> 5, lane = threadIdx.x & 31;
  int nl = lane & 15, hs = lane >> 4;
  int m0 = wave * 16;

#if USE_TDM
  if (threadIdx.x < 32) {            // one wave drives the Tensor Data Mover
    tdm_copy_f16(Qs, qh,  NTOK * HD);
    tdm_copy_f16(Ks, kh,  NTOK * HD);
    tdm_copy_f16(Vs, vth, NTOK * HD);
    __builtin_amdgcn_s_wait_tensorcnt((short)0);
  }
  __syncthreads();
#else
  {
    int t = threadIdx.x;
#pragma unroll
    for (int u = 0; u < 2; ++u) {
      ((v8h*)Qs)[t * 2 + u] = ((const v8h*)qh)[t * 2 + u];
      ((v8h*)Ks)[t * 2 + u] = ((const v8h*)kh)[t * 2 + u];
      ((v8h*)Vs)[t * 2 + u] = ((const v8h*)vth)[t * 2 + u];
    }
  }
  __syncthreads();
#endif

  // S = (Q*scale) @ K^T : 16x64 per wave, operands from LDS
  v16h aq = load_a16(Qs + m0 * HD, HD);
  v8f s[4];
#pragma unroll
  for (int ct = 0; ct < 4; ++ct) {
    v16h b = load_bt16(Ks + ct * 16 * HD, HD);
    v8f z = {};
    s[ct] = wmma16(aq, b, z);
  }
#pragma unroll
  for (int ct = 0; ct < 4; ++ct)
#pragma unroll
    for (int r = 0; r < 8; ++r) {
      int mm = m0 + hs * 8 + r, nn = ct * 16 + nl;
      s[ct][r] = s[ct][r] + bb[mm * 64 + nn];
    }
  // softmax over 64 cols: each row lives in one 16-lane half across 4 tiles
#pragma unroll
  for (int r = 0; r < 8; ++r) {
    float mx = fmaxf(fmaxf(s[0][r], s[1][r]), fmaxf(s[2][r], s[3][r]));
    for (int off = 8; off >= 1; off >>= 1) mx = fmaxf(mx, __shfl_xor(mx, off, 32));
    float sm = 0.f;
#pragma unroll
    for (int ct = 0; ct < 4; ++ct) { s[ct][r] = __expf(s[ct][r] - mx); sm += s[ct][r]; }
    for (int off = 8; off >= 1; off >>= 1) sm += __shfl_xor(sm, off, 32);
    float inv = 1.0f / sm;
    int mm = m0 + hs * 8 + r;
#pragma unroll
    for (int ct = 0; ct < 4; ++ct)
      P[mm][ct * 16 + nl] = (_Float16)(s[ct][r] * inv);
  }
  __syncthreads();

  // O = P @ V : 16x32 per wave, K=64 as two chained WMMAs; B from V^T in LDS
  v8f o0 = {}, o1 = {};
#pragma unroll
  for (int k0 = 0; k0 < 64; k0 += 32) {
    v16h ap = load_a16(&P[m0][k0], 64);
    v16h b0 = load_bt16(Vs + 0  * NTOK + k0, NTOK);
    v16h b1 = load_bt16(Vs + 16 * NTOK + k0, NTOK);
    o0 = wmma16(ap, b0, o0);
    o1 = wmma16(ap, b1, o1);
  }
#pragma unroll
  for (int r = 0; r < 8; ++r) {
    int mm = m0 + hs * 8 + r;
    size_t row = (size_t)(win * NTOK + mm) * 256 + head * HD;
    o[row + nl]      = (_Float16)o0[r];
    o[row + 16 + nl] = (_Float16)o1[r];
  }
}

// ---------------- GEMM: proj + window-reverse + residual1 (hoisted) ----------------

__global__ void __launch_bounds__(128) k_gemm_proj(const _Float16* __restrict__ ob,
                                                   const _Float16* __restrict__ wpT,
                                                   const float* __restrict__ pb,
                                                   const float* __restrict__ x,
                                                   float* __restrict__ x1) {
  int wave = threadIdx.x >> 5, lane = threadIdx.x & 31;
  int n0 = blockIdx.x * 128 + wave * 32;
  int m0 = blockIdx.y * 64;
  v8f acc[4][2] = {};
  gemm_tile_64x32(ob + (size_t)m0 * 256, 256, wpT + (size_t)n0 * 256, 256, 256, acc);
  int nl = lane & 15, hs = lane >> 4;
  int win = m0 >> 6;                 // block-uniform
  int b_ = win >> 8, wh = (win >> 4) & 15, wwi = win & 15;
  int pb0 = wh * WS * WW + wwi * WS; // spatial base of this window
#pragma unroll
  for (int j = 0; j < 2; ++j) {
    int nn = n0 + j * 16 + nl;
    float bias = pb[nn];
    const float* xb = x + ((size_t)(b_ * 256 + nn)) * HWSZ + pb0;
    float* x1b = x1 + ((size_t)(b_ * HWSZ + pb0)) * 256 + nn;
#pragma unroll
    for (int i = 0; i < 4; ++i)
#pragma unroll
      for (int r = 0; r < 8; ++r) {
        int t = i * 16 + hs * 8 + r;
        int po = (t >> 3) * WW + (t & 7);
        x1b[(size_t)po * 256] = xb[po] + acc[i][j][r] + bias;
      }
  }
}

// ---------------- LN2: x1 -> xn2 f16 ----------------

__global__ void __launch_bounds__(256) k_ln2(const float* __restrict__ x1,
                                             const float* __restrict__ g,
                                             const float* __restrict__ bta,
                                             _Float16* __restrict__ xn2) {
  int row = blockIdx.x * 8 + (threadIdx.x >> 5);
  int lane = threadIdx.x & 31;
  const float* xr = x1 + (size_t)row * 256;
  float vals[8], sum = 0.f;
#pragma unroll
  for (int j = 0; j < 8; ++j) { vals[j] = xr[lane + 32 * j]; sum += vals[j]; }
  for (int off = 16; off >= 1; off >>= 1) sum += __shfl_xor(sum, off, 32);
  float mu = sum * (1.f / 256.f);
  float var = 0.f;
#pragma unroll
  for (int j = 0; j < 8; ++j) { float d = vals[j] - mu; var += d * d; }
  for (int off = 16; off >= 1; off >>= 1) var += __shfl_xor(var, off, 32);
  float rs = rsqrtf(var * (1.f / 256.f) + 1e-5f);
  _Float16* yr = xn2 + (size_t)row * 256;
#pragma unroll
  for (int j = 0; j < 8; ++j) {
    int c = lane + 32 * j;
    yr[c] = (_Float16)((vals[j] - mu) * rs * g[c] + bta[c]);
  }
}

// ---------------- GEMM: MLP fc1 + exact GELU (hoisted) ----------------

__global__ void __launch_bounds__(128) k_gemm_mlp1(const _Float16* __restrict__ xn2,
                                                   const _Float16* __restrict__ w1T,
                                                   const float* __restrict__ b1,
                                                   _Float16* __restrict__ h1) {
  int wave = threadIdx.x >> 5, lane = threadIdx.x & 31;
  int n0 = blockIdx.x * 128 + wave * 32;
  int m0 = blockIdx.y * 64;
  v8f acc[4][2] = {};
  gemm_tile_64x32(xn2 + (size_t)m0 * 256, 256, w1T + (size_t)n0 * 256, 256, 256, acc);
  int nl = lane & 15, hs = lane >> 4;
#pragma unroll
  for (int j = 0; j < 2; ++j) {
    int nn = n0 + j * 16 + nl;
    float bias = b1[nn];
    _Float16* hb = h1 + (size_t)m0 * HID + nn;
#pragma unroll
    for (int i = 0; i < 4; ++i)
#pragma unroll
      for (int r = 0; r < 8; ++r) {
        int t = i * 16 + hs * 8 + r;
        float vv = acc[i][j][r] + bias;
        hb[(size_t)t * HID] = (_Float16)(0.5f * vv * (1.0f + erff(vv * 0.70710678118654752f)));
      }
  }
}

// ---------------- GEMM: MLP fc2 + residual2 + NCHW scatter (hoisted) ----------------

__global__ void __launch_bounds__(128) k_gemm_mlp2(const _Float16* __restrict__ h1,
                                                   const _Float16* __restrict__ w2T,
                                                   const float* __restrict__ b2,
                                                   const float* __restrict__ x1,
                                                   float* __restrict__ out) {
  int wave = threadIdx.x >> 5, lane = threadIdx.x & 31;
  int n0 = blockIdx.x * 128 + wave * 32;
  int m0 = blockIdx.y * 64;
  v8f acc[4][2] = {};
  gemm_tile_64x32(h1 + (size_t)m0 * HID, HID, w2T + (size_t)n0 * HID, HID, HID, acc);
  int nl = lane & 15, hs = lane >> 4;
  int b_ = m0 >> 14;                 // block-uniform (64 | 16384)
  int pb0 = m0 & (HWSZ - 1);
#pragma unroll
  for (int j = 0; j < 2; ++j) {
    int nn = n0 + j * 16 + nl;
    float bias = b2[nn];
    const float* x1b = x1 + (size_t)m0 * 256 + nn;
    float* outb = out + ((size_t)(b_ * 256 + nn)) * HWSZ + pb0;
#pragma unroll
    for (int i = 0; i < 4; ++i)
#pragma unroll
      for (int r = 0; r < 8; ++r) {
        int t = i * 16 + hs * 8 + r;
        outb[t] = acc[i][j][r] + bias + x1b[(size_t)t * 256];
      }
  }
}

// ---------------- launcher ----------------

extern "C" void kernel_launch(void* const* d_in, const int* in_sizes, int n_in,
                              void* d_out, int out_size, void* d_ws, size_t ws_size,
                              hipStream_t stream) {
  const float* x      = (const float*)d_in[0];
  const float* ln1_g  = (const float*)d_in[1];
  const float* ln1_b  = (const float*)d_in[2];
  const float* qkv_w  = (const float*)d_in[3];
  const float* qkv_b  = (const float*)d_in[4];
  const float* proj_w = (const float*)d_in[5];
  const float* proj_b = (const float*)d_in[6];
  const float* rpb    = (const float*)d_in[7];
  const float* ln2_g  = (const float*)d_in[8];
  const float* ln2_b  = (const float*)d_in[9];
  const float* w1     = (const float*)d_in[10];
  const float* b1     = (const float*)d_in[11];
  const float* w2     = (const float*)d_in[12];
  const float* b2     = (const float*)d_in[13];
  const int*   rpi    = (const int*)d_in[14];
  float* out = (float*)d_out;

  const size_t MiB = (size_t)1 << 20;
  char* w = (char*)d_ws;
  _Float16* xw  = (_Float16*)(w);                 // 64 MiB  [MROWS][256]
  _Float16* qb  = (_Float16*)(w + 64  * MiB);     // 64 MiB
  _Float16* kb  = (_Float16*)(w + 128 * MiB);     // 64 MiB
  _Float16* vtb = (_Float16*)(w + 192 * MiB);     // 64 MiB  (V transposed per bh)
  _Float16* ob  = xw;                             // alias: xw dead after QKV
  float*    x1  = (float*)(w + 256 * MiB);        // 128 MiB [MROWS][256] fp32
  _Float16* xn2 = qb;                             // alias: q dead after attn
  _Float16* h1  = kb;                             // alias: k+v span = 128 MiB
  char* wex = w + 384 * MiB;
  _Float16* wqT = (_Float16*)wex;                 // [768][256]
  _Float16* wpT = wqT + 196608;                   // [256][256]
  _Float16* w1T = wpT + 65536;                    // [512][256]
  _Float16* w2T = w1T + 131072;                   // [256][512]
  float*    bd  = (float*)(w2T + 131072);         // [8][4096] fp32

  k_transpose_f16<<<768, 256, 0, stream>>>(qkv_w, wqT, 256, 768);
  k_transpose_f16<<<256, 256, 0, stream>>>(proj_w, wpT, 256, 256);
  k_transpose_f16<<<512, 256, 0, stream>>>(w1, w1T, 256, 512);
  k_transpose_f16<<<512, 256, 0, stream>>>(w2, w2T, 512, 256);
  k_biasdense<<<128, 256, 0, stream>>>(rpb, rpi, bd);
  k_ln1<<<dim3(HWSZ / 16, BATCH), 256, 0, stream>>>(x, ln1_g, ln1_b, xw);
  k_gemm_qkv<<<dim3(768 / 128, MROWS / 64), 128, 0, stream>>>(xw, wqT, qkv_b, qb, kb, vtb);
  k_attn<<<NWIN * NH, 128, 0, stream>>>(qb, kb, vtb, bd, ob);
  k_gemm_proj<<<dim3(256 / 128, MROWS / 64), 128, 0, stream>>>(ob, wpT, proj_b, x, x1);
  k_ln2<<<MROWS / 8, 256, 0, stream>>>(x1, ln2_g, ln2_b, xn2);
  k_gemm_mlp1<<<dim3(HID / 128, MROWS / 64), 128, 0, stream>>>(xn2, w1T, b1, h1);
  k_gemm_mlp2<<<dim3(256 / 128, MROWS / 64), 128, 0, stream>>>(h1, w2T, b2, x1, out);
  (void)in_sizes; (void)n_in; (void)out_size; (void)ws_size;
}
